// CausalSelfAttention_22359599743131
// MI455X (gfx1250) — matrix operand order repaired and checked by reading, not compiled, not run on hardware
//
#include <hip/hip_runtime.h>
#include <hip/hip_bf16.h>

// ---------------------------------------------------------------------------
// Causal self-attention fwd for MI455X (gfx1250, wave32, WMMA 16x16x32 f16).
// B=1, T=4096, C=512, H=8, HS=64.
// All GEMM stages run through v_wmma_f32_16x16x32_f16 with f32 accumulation;
// 32x64 output tiles per wave so each B fragment feeds two WMMAs.
// ---------------------------------------------------------------------------

typedef __attribute__((ext_vector_type(16))) _Float16 v16h;
typedef __attribute__((ext_vector_type(8)))  _Float16 v8h;
typedef __attribute__((ext_vector_type(8)))  float    v8f;

#define TT  4096
#define CC  512
#define HH  8
#define HS  64

__device__ __forceinline__ v8f wmma_f16(v16h a, v16h b, v8f c) {
  return __builtin_amdgcn_wmma_f32_16x16x32_f16(false, a, false, b,
                                                (short)0, c, false, false);
}

// A-matrix fragment: 16x32 (MxK), row-major source with leading dim ld.
// Lane L<16 : halves 0..7 = A[L][k0..k0+7],    halves 8..15 = A[L][k0+16..23]
// Lane L>=16: halves 0..7 = A[L-16][k0+8..15], halves 8..15 = A[L-16][k0+24..31]
__device__ __forceinline__ v16h load_frag_a(const _Float16* base, int ld,
                                            int row0, int k0, int lane) {
  int r = lane & 15, hi = lane >> 4;
  const _Float16* p = base + (long)(row0 + r) * ld + k0 + hi * 8;
  v16h a;
  ((v8h*)&a)[0] = *(const v8h*)(p);
  ((v8h*)&a)[1] = *(const v8h*)(p + 16);
  return a;
}

// B-matrix fragment 32x16 (KxN) where B[K][N] = W[n0+N][k0+K] (W row-major,
// leading dim ld) -> same k order as A: k0+hi*8+0..7, k0+16+hi*8+0..7.
__device__ __forceinline__ v16h load_frag_bT(const _Float16* base, int ld,
                                             int n0, int k0, int lane) {
  int r = lane & 15, hi = lane >> 4;
  const _Float16* p = base + (long)(n0 + r) * ld + k0 + hi * 8;
  v16h b;
  ((v8h*)&b)[0] = *(const v8h*)(p);
  ((v8h*)&b)[1] = *(const v8h*)(p + 16);
  return b;
}

// ---------------------------------------------------------------------------
// f32 -> f16 conversion
// ---------------------------------------------------------------------------
__global__ void cvt_f32_f16(const float* __restrict__ in,
                            _Float16* __restrict__ out, int n) {
  int i = blockIdx.x * blockDim.x + threadIdx.x;
  if (i < n) out[i] = (_Float16)in[i];
}

// ---------------------------------------------------------------------------
// C[M][N] (f32) = A[M][K] (f16) @ W[N][K]^T (f16). One wave -> 32x64 tile.
// Each B fragment is consumed by two WMMAs (two A row sub-tiles).
// ---------------------------------------------------------------------------
__global__ __launch_bounds__(32) void gemm_f16_wmma(
    const _Float16* __restrict__ A, const _Float16* __restrict__ W,
    float* __restrict__ Cm, int M, int N, int K) {
  int lane = threadIdx.x & 31;
  int tiles_n = N >> 6;
  int row0 = (blockIdx.x / tiles_n) * 32;
  int col0 = (blockIdx.x % tiles_n) * 64;
  int r = lane & 15, hi = lane >> 4;
  v8f acc[2][4] = {};
  for (int k0 = 0; k0 < K; k0 += 32) {
    // speculative prefetch of next K-chunk (global_prefetch_b8; OOB dropped)
    __builtin_prefetch(A + (long)(row0 + r) * K + k0 + 32, 0, 1);
    __builtin_prefetch(W + (long)(col0 + r) * K + k0 + 32, 0, 1);
    v16h a0 = load_frag_a(A, K, row0, k0, lane);
    v16h a1 = load_frag_a(A, K, row0 + 16, k0, lane);
#pragma unroll
    for (int c = 0; c < 4; ++c) {
      v16h b = load_frag_bT(W, K, col0 + c * 16, k0, lane);
      acc[0][c] = wmma_f16(a0, b, acc[0][c]);
      acc[1][c] = wmma_f16(a1, b, acc[1][c]);
    }
  }
#pragma unroll
  for (int i = 0; i < 2; ++i)
#pragma unroll
    for (int c = 0; c < 4; ++c)
#pragma unroll
      for (int j = 0; j < 8; ++j)
        Cm[(long)(row0 + i * 16 + j + hi * 8) * N + col0 + c * 16 + r] =
            acc[i][c][j];
}

// ---------------------------------------------------------------------------
// RoPE on q,k + scatter:  qkv (T x 1536 f32) ->
//   qh, kh : (H, T, HS) f16 (rotated);  vT : (H, HS, T) f16 (transposed)
// ---------------------------------------------------------------------------
__global__ void rope_scatter(const float* __restrict__ qkv,
                             const float* __restrict__ rope,
                             _Float16* __restrict__ qh,
                             _Float16* __restrict__ kh,
                             _Float16* __restrict__ vT) {
  int idx = blockIdx.x * blockDim.x + threadIdx.x;
  if (idx >= TT * HH * (HS / 2)) return;
  int i = idx & 31;
  int h = (idx >> 5) & 7;
  int t = idx >> 8;
  float cs = rope[t * HS + 2 * i];
  float sn = rope[t * HS + 2 * i + 1];
  const float* q = qkv + (long)t * (3 * CC) + h * HS;
  const float* k = q + CC;
  const float* v = q + 2 * CC;
  float q0 = q[2 * i], q1 = q[2 * i + 1];
  float k0 = k[2 * i], k1 = k[2 * i + 1];
  long qb = (long)h * TT * HS + (long)t * HS;
  qh[qb + 2 * i]     = (_Float16)(q0 * cs - q1 * sn);
  qh[qb + 2 * i + 1] = (_Float16)(q1 * cs + q0 * sn);
  kh[qb + 2 * i]     = (_Float16)(k0 * cs - k1 * sn);
  kh[qb + 2 * i + 1] = (_Float16)(k1 * cs + k0 * sn);
  long vb = (long)h * HS * TT;
  vT[vb + (long)(2 * i) * TT + t]     = (_Float16)v[2 * i];
  vT[vb + (long)(2 * i + 1) * TT + t] = (_Float16)v[2 * i + 1];
}

// ---------------------------------------------------------------------------
// Flash attention: one wave per (head, 32-row q tile). Online softmax.
// K/V fragments are shared by the two 16-row q sub-tiles (2x WMMA per load).
// yh out: (T, C) f16, y[t][h*HS + d].
// ---------------------------------------------------------------------------
__global__ __launch_bounds__(32) void attn_fwd(
    const _Float16* __restrict__ qh, const _Float16* __restrict__ kh,
    const _Float16* __restrict__ vT, _Float16* __restrict__ yh) {
  __shared__ _Float16 Pl[32 * 32];  // P tile staging (C-layout -> A-layout)
  int lane = threadIdx.x & 31;
  int r = lane & 15, hi = lane >> 4;
  int qt = blockIdx.x % (TT / 32);
  int h  = blockIdx.x / (TT / 32);
  int q0 = qt * 32;

  const _Float16* Q  = qh + (long)h * TT * HS;
  const _Float16* Km = kh + (long)h * TT * HS;
  const _Float16* Vt = vT + (long)h * HS * TT;

  // Q fragments: two 16-row sub-tiles x two 32-wide d-chunks (loop-resident)
  v16h aq[2][2];
#pragma unroll
  for (int a = 0; a < 2; ++a) {
    aq[a][0] = load_frag_a(Q, HS, q0 + a * 16, 0, lane);
    aq[a][1] = load_frag_a(Q, HS, q0 + a * 16, 32, lane);
  }

  v8f o[2][4] = {};
  float m_i[2][8], l_i[2][8];
#pragma unroll
  for (int a = 0; a < 2; ++a)
#pragma unroll
    for (int j = 0; j < 8; ++j) { m_i[a][j] = -1e30f; l_i[a][j] = 0.0f; }

  for (int kt0 = 0; kt0 < q0 + 32; kt0 += 32) {
    // ---- scores: two 16-col k sub-tiles; K-frags reused by both q tiles ---
    v8f s[2][2];  // [q sub-tile][k sub-tile]
#pragma unroll
    for (int sub = 0; sub < 2; ++sub) {
      int kc0 = kt0 + sub * 16;
      v16h b0 = load_frag_bT(Km, HS, kc0, 0, lane);   // k^T, d = 0..31
      v16h b1 = load_frag_bT(Km, HS, kc0, 32, lane);  // k^T, d = 32..63
#pragma unroll
      for (int a = 0; a < 2; ++a) {
        v8f sv = {};
        sv = wmma_f16(aq[a][0], b0, sv);
        sv = wmma_f16(aq[a][1], b1, sv);
#pragma unroll
        for (int j = 0; j < 8; ++j) {
          int row = q0 + a * 16 + j + hi * 8;
          int col = kc0 + r;
          s[a][sub][j] = (col <= row) ? sv[j] * 0.125f : -1e30f;  // 1/sqrt(64)
        }
      }
    }
    // ---- online softmax per q sub-tile (rows live in 16-lane halves) -----
#pragma unroll
    for (int a = 0; a < 2; ++a) {
#pragma unroll
      for (int j = 0; j < 8; ++j) {
        float mt = fmaxf(s[a][0][j], s[a][1][j]);
#pragma unroll
        for (int m = 1; m <= 8; m <<= 1) mt = fmaxf(mt, __shfl_xor(mt, m, 32));
        float m_new = fmaxf(m_i[a][j], mt);
        float alpha = __expf(m_i[a][j] - m_new);
        m_i[a][j] = m_new;
        float p0 = __expf(s[a][0][j] - m_new);
        float p1 = __expf(s[a][1][j] - m_new);
        float rs = p0 + p1;
#pragma unroll
        for (int m = 1; m <= 8; m <<= 1) rs += __shfl_xor(rs, m, 32);
        l_i[a][j] = l_i[a][j] * alpha + rs;
#pragma unroll
        for (int c = 0; c < 4; ++c) o[a][c][j] *= alpha;
        Pl[(a * 16 + j + hi * 8) * 32 + r]      = (_Float16)p0;
        Pl[(a * 16 + j + hi * 8) * 32 + 16 + r] = (_Float16)p1;
      }
    }
    // ---- P (2 x 16x32) as A fragments from LDS ----
    v16h ap[2];
#pragma unroll
    for (int a = 0; a < 2; ++a) {
      const _Float16* p = Pl + (a * 16 + r) * 32 + hi * 8;
      ((v8h*)&ap[a])[0] = *(const v8h*)(p);
      ((v8h*)&ap[a])[1] = *(const v8h*)(p + 16);
    }
    // ---- O += P @ V; each V fragment feeds both q sub-tiles ----
#pragma unroll
    for (int c = 0; c < 4; ++c) {
      v16h bv;
      const _Float16* p = Vt + (long)(c * 16 + r) * TT + kt0 + hi * 8;
      ((v8h*)&bv)[0] = *(const v8h*)(p);
      ((v8h*)&bv)[1] = *(const v8h*)(p + 16);
      o[0][c] = wmma_f16(ap[0], bv, o[0][c]);
      o[1][c] = wmma_f16(ap[1], bv, o[1][c]);
    }
  }
  // ---- normalize and emit y (f16, row-major T x C) ----
#pragma unroll
  for (int a = 0; a < 2; ++a)
#pragma unroll
    for (int c = 0; c < 4; ++c)
#pragma unroll
      for (int j = 0; j < 8; ++j)
        yh[(long)(q0 + a * 16 + j + hi * 8) * CC + h * HS + c * 16 + r] =
            (_Float16)(o[a][c][j] / l_i[a][j]);
}

// ---------------------------------------------------------------------------
// Host-side launch
// ---------------------------------------------------------------------------
extern "C" void kernel_launch(void* const* d_in, const int* in_sizes, int n_in,
                              void* d_out, int out_size, void* d_ws,
                              size_t ws_size, hipStream_t stream) {
  const float* x      = (const float*)d_in[0];  // (1,T,C)
  const float* rope   = (const float*)d_in[1];  // (T,32,2)
  // d_in[2] = mask (bool) -- causality computed analytically, unused
  const float* W_attn = (const float*)d_in[3];  // (3C,C)
  const float* W_proj = (const float*)d_in[4];  // (C,C)
  float* out = (float*)d_out;

  char* ws = (char*)d_ws;
  size_t off = 0;
  auto alloc = [&](size_t bytes) {
    char* p = ws + off;
    off = (off + bytes + 255) & ~(size_t)255;
    return p;
  };
  _Float16* xh  = (_Float16*)alloc((size_t)TT * CC * 2);
  _Float16* Wah = (_Float16*)alloc((size_t)3 * CC * CC * 2);
  _Float16* Wph = (_Float16*)alloc((size_t)CC * CC * 2);
  float*    qkv = (float*)   alloc((size_t)TT * 3 * CC * 4);
  _Float16* qh  = (_Float16*)alloc((size_t)HH * TT * HS * 2);
  _Float16* kh  = (_Float16*)alloc((size_t)HH * TT * HS * 2);
  _Float16* vT  = (_Float16*)alloc((size_t)HH * HS * TT * 2);
  _Float16* yh  = (_Float16*)alloc((size_t)TT * CC * 2);
  (void)ws_size; (void)in_sizes; (void)n_in; (void)out_size;

  // 1) convert to f16
  {
    int n1 = TT * CC, n2 = 3 * CC * CC, n3 = CC * CC;
    cvt_f32_f16<<<(n1 + 255) / 256, 256, 0, stream>>>(x, xh, n1);
    cvt_f32_f16<<<(n2 + 255) / 256, 256, 0, stream>>>(W_attn, Wah, n2);
    cvt_f32_f16<<<(n3 + 255) / 256, 256, 0, stream>>>(W_proj, Wph, n3);
  }
  // 2) qkv = x @ W_attn^T   (4096x512 @ 512x1536)
  gemm_f16_wmma<<<(TT / 32) * (3 * CC / 64), 32, 0, stream>>>(
      xh, Wah, qkv, TT, 3 * CC, CC);
  // 3) RoPE + scatter to head-major layouts (V transposed)
  {
    int n = TT * HH * (HS / 2);
    rope_scatter<<<(n + 255) / 256, 256, 0, stream>>>(qkv, rope, qh, kh, vT);
  }
  // 4) flash attention (one wave per head x 32-row q tile)
  attn_fwd<<<HH * (TT / 32), 32, 0, stream>>>(qh, kh, vT, yh);
  // 5) out = y @ W_proj^T
  gemm_f16_wmma<<<(TT / 32) * (CC / 64), 32, 0, stream>>>(
      yh, Wph, out, TT, CC, CC);
}
